// MyModel_44624710205749
// MI455X (gfx1250) — matrix operand order repaired and checked by reading, not compiled, not run on hardware
//
#include <hip/hip_runtime.h>
#include <hip/hip_bf16.h>

// ---------------------------------------------------------------------------
// Problem constants (match reference)
// ---------------------------------------------------------------------------
#define N_ROWS 32768
#define P_DIM  1024
#define HD_DIM 1024
#define D_DIM  2048   // P + HD

typedef __bf16        v16bf __attribute__((ext_vector_type(16)));
typedef float         v8f   __attribute__((ext_vector_type(8)));
typedef unsigned int  v8u   __attribute__((ext_vector_type(8)));

// round-to-nearest-even fp32 -> bf16 (upper 16 bits)
__device__ __forceinline__ unsigned int bf16_hi_rne(float f) {
    unsigned int u = __float_as_uint(f);
    return (u + 0x7fffu + ((u >> 16) & 1u)) >> 16;
}

// split fp32 into bf16 hi + bf16 residual lo (captures ~16 mantissa bits)
__device__ __forceinline__ void split_bf16(float f, unsigned int &hi, unsigned int &lo) {
    hi = bf16_hi_rne(f);
    float fh = __uint_as_float(hi << 16);
    lo = bf16_hi_rne(f - fh);
}

// ---------------------------------------------------------------------------
// Fused GEMM + bias + ReLU:  C[row, col] = relu( sum_k A[row,k]*W[col,k] + b[col] )
// A is virtually concat(A0, A1) along k at `split` (GEMM1: [x|h]; GEMM2: a1).
// Tile: 128x128 per workgroup, BK=32, 256 threads = 8 waves, wave tile 64x32.
// Split-precision bf16 WMMA: 3x v_wmma_f32_16x16x32_bf16 per 32-k step.
// ---------------------------------------------------------------------------
__global__ __launch_bounds__(256)
void gemm_relu_kernel(const float* __restrict__ A0, const float* __restrict__ A1,
                      int lda0, int lda1, int split,
                      const float* __restrict__ W, const float* __restrict__ bias,
                      float* __restrict__ C, int K, int ldc)
{
    // hi/lo bf16 planes in LDS; row stride 34 (even -> 4B-aligned packed reads)
    __shared__ unsigned short As_hi[128 * 34];
    __shared__ unsigned short As_lo[128 * 34];
    __shared__ unsigned short Ws_hi[128 * 34];
    __shared__ unsigned short Ws_lo[128 * 34];

    const int tid     = threadIdx.x;
    const int lane    = tid & 31;
    const int w       = tid >> 5;
    const int waveR   = (w >> 2) * 64;   // 2 wave-rows
    const int waveC   = (w & 3) * 32;    // 4 wave-cols
    const int rowBase = blockIdx.y * 128;
    const int colBase = blockIdx.x * 128;
    const int hsel    = (lane < 16) ? 0 : 1;
    const int l15     = lane & 15;

    const v8f vzero = {0.f, 0.f, 0.f, 0.f, 0.f, 0.f, 0.f, 0.f};
    v8f acc[4][2];
#pragma unroll
    for (int r = 0; r < 4; ++r)
#pragma unroll
        for (int c = 0; c < 2; ++c) acc[r][c] = vzero;

    for (int kb = 0; kb < K; kb += 32) {
        // ---- stage 128x32 tiles of A and W into LDS as bf16 hi/lo planes ----
#pragma unroll
        for (int i = 0; i < 4; ++i) {
            int q   = tid + i * 256;       // 1024 float4's per tile
            int r   = q >> 3;              // 0..127
            int col = (q & 7) * 4;         // 0..28 step 4
            int o   = r * 34 + col;

            // A side (virtual concat at `split`; kb,split multiples of 32 -> no straddle)
            int kg = kb + col;
            const float* srcA = (kg < split)
                ? (A0 + (size_t)(rowBase + r) * lda0 + kg)
                : (A1 + (size_t)(rowBase + r) * lda1 + (kg - split));
            float4 va = *(const float4*)srcA;
            if (kb + 32 < K) __builtin_prefetch(srcA + 32, 0, 1);
            unsigned int h0, l0, h1, l1, h2, l2, h3, l3;
            split_bf16(va.x, h0, l0); split_bf16(va.y, h1, l1);
            split_bf16(va.z, h2, l2); split_bf16(va.w, h3, l3);
            *(unsigned int*)&As_hi[o]     = h0 | (h1 << 16);
            *(unsigned int*)&As_hi[o + 2] = h2 | (h3 << 16);
            *(unsigned int*)&As_lo[o]     = l0 | (l1 << 16);
            *(unsigned int*)&As_lo[o + 2] = l2 | (l3 << 16);

            // W side: W is [Mcols, K] row-major
            const float* srcW = W + (size_t)(colBase + r) * K + kb + col;
            float4 vw = *(const float4*)srcW;
            if (kb + 32 < K) __builtin_prefetch(srcW + 32, 0, 1);
            split_bf16(vw.x, h0, l0); split_bf16(vw.y, h1, l1);
            split_bf16(vw.z, h2, l2); split_bf16(vw.w, h3, l3);
            *(unsigned int*)&Ws_hi[o]     = h0 | (h1 << 16);
            *(unsigned int*)&Ws_hi[o + 2] = h2 | (h3 << 16);
            *(unsigned int*)&Ws_lo[o]     = l0 | (l1 << 16);
            *(unsigned int*)&Ws_lo[o + 2] = l2 | (l3 << 16);
        }
        __syncthreads();

        // ---- B fragments (32x16 KxN): V0..3 lanes0-15 K=0..7, lanes16-31 K=8..15; V4..7 +16
        v16bf Bh[2], Bl[2];
#pragma unroll
        for (int c = 0; c < 2; ++c) {
            int m = waveC + c * 16 + l15;   // output column within tile
            v8u bh, bl;
#pragma unroll
            for (int v = 0; v < 8; ++v) {
                int k = ((v < 4) ? 2 * v : 16 + 2 * (v - 4)) + hsel * 8;
                bh[v] = *(const unsigned int*)&Ws_hi[m * 34 + k];
                bl[v] = *(const unsigned int*)&Ws_lo[m * 34 + k];
            }
            Bh[c] = __builtin_bit_cast(v16bf, bh);
            Bl[c] = __builtin_bit_cast(v16bf, bl);
        }

        // ---- A fragments (16x32 MxK) + 3 WMMAs per (r,c) ----
#pragma unroll
        for (int r = 0; r < 4; ++r) {
            int rowa = waveR + r * 16 + l15;
            v8u ah, al;
#pragma unroll
            for (int v = 0; v < 8; ++v) {
                int k = ((v < 4) ? 2 * v : 16 + 2 * (v - 4)) + hsel * 8;
                ah[v] = *(const unsigned int*)&As_hi[rowa * 34 + k];
                al[v] = *(const unsigned int*)&As_lo[rowa * 34 + k];
            }
            v16bf Ah = __builtin_bit_cast(v16bf, ah);
            v16bf Al = __builtin_bit_cast(v16bf, al);
#pragma unroll
            for (int c = 0; c < 2; ++c) {
                acc[r][c] = __builtin_amdgcn_wmma_f32_16x16x32_bf16(
                    false, Ah, false, Bh[c], (short)0, acc[r][c], false, false);
                acc[r][c] = __builtin_amdgcn_wmma_f32_16x16x32_bf16(
                    false, Ah, false, Bl[c], (short)0, acc[r][c], false, false);
                acc[r][c] = __builtin_amdgcn_wmma_f32_16x16x32_bf16(
                    false, Al, false, Bh[c], (short)0, acc[r][c], false, false);
            }
        }
        __syncthreads();
    }

    // ---- epilogue: bias + relu + store (C/D layout: lane= N, vgpr(+8 for hi half)= M)
#pragma unroll
    for (int c = 0; c < 2; ++c) {
        int colg = colBase + waveC + c * 16 + l15;
        float bv = bias[colg];
#pragma unroll
        for (int r = 0; r < 4; ++r) {
#pragma unroll
            for (int j = 0; j < 8; ++j) {
                int rowg = rowBase + waveR + r * 16 + j + hsel * 8;
                float v = acc[r][c][j] + bv;
                v = v > 0.f ? v : 0.f;
                C[(size_t)rowg * ldc + colg] = v;
            }
        }
    }
}

// ---------------------------------------------------------------------------
// Reduction / loss kernels (bandwidth-trivial vs. the GEMMs)
// ---------------------------------------------------------------------------
__global__ __launch_bounds__(1024)
void zero_kernel(float* musum, float* scal) {
    int t = threadIdx.x;
    musum[t] = 0.f;
    if (t < 3) scal[t] = 0.f;
}

// column sums of h -> musum (atomic accumulate over row chunks)
__global__ __launch_bounds__(256)
void colsum_kernel(const float* __restrict__ h, float* __restrict__ musum) {
    int col = blockIdx.x * 256 + threadIdx.x;
    int r0  = blockIdx.y * 1024;
    float s = 0.f;
    for (int i = 0; i < 1024; ++i)
        s += h[(size_t)(r0 + i) * HD_DIM + col];
    atomicAdd(&musum[col], s);
}

// sum over rows 0..N-2 of (sum_j h[i,j])^2  -> scal[0]
__global__ __launch_bounds__(256)
void rowsq_kernel(const float* __restrict__ h, float* __restrict__ ssq) {
    int w = threadIdx.x >> 5, lane = threadIdx.x & 31;
    int row = blockIdx.x * 8 + w;
    if (row >= N_ROWS - 1) return;
    const float* hr = h + (size_t)row * HD_DIM;
    float s = 0.f;
    for (int j = lane; j < HD_DIM; j += 32) s += hr[j];
    for (int off = 16; off > 0; off >>= 1) s += __shfl_down(s, off, 32);
    if (lane == 0) atomicAdd(ssq, s * s);
}

// fused ||f[:-1] - (x[1:]-x[:-1])||^2 -> scal[1], ||h[1:]-h[:-1]||^2 -> scal[2]
__global__ __launch_bounds__(256)
void diff_kernel(const float* __restrict__ f, const float* __restrict__ x,
                 const float* __restrict__ h, float* __restrict__ sx,
                 float* __restrict__ sh) {
    __shared__ float rx[256], rh[256];
    const size_t total = (size_t)(N_ROWS - 1) * P_DIM;
    float ax = 0.f, ah = 0.f;
    for (size_t idx = (size_t)blockIdx.x * 256 + threadIdx.x; idx < total;
         idx += (size_t)gridDim.x * 256) {
        float dx = f[idx] - (x[idx + P_DIM] - x[idx]);
        ax += dx * dx;
        float dh = h[idx + HD_DIM] - h[idx];
        ah += dh * dh;
    }
    rx[threadIdx.x] = ax; rh[threadIdx.x] = ah;
    __syncthreads();
    for (int s = 128; s > 0; s >>= 1) {
        if (threadIdx.x < s) {
            rx[threadIdx.x] += rx[threadIdx.x + s];
            rh[threadIdx.x] += rh[threadIdx.x + s];
        }
        __syncthreads();
    }
    if (threadIdx.x == 0) { atomicAdd(sx, rx[0]); atomicAdd(sh, rh[0]); }
}

// finalize: rank-2 analytic slogdet + losses (fp64 scalar math on one thread)
__global__ __launch_bounds__(1024)
void finalize_kernel(const float* __restrict__ musum, const float* __restrict__ scal,
                     float* __restrict__ out) {
    __shared__ double s1[1024], s2[1024];
    int t = threadIdx.x;
    double mu = (double)musum[t] / (double)N_ROWS;
    s1[t] = mu; s2[t] = mu * mu;
    __syncthreads();
    for (int s = 512; s > 0; s >>= 1) {
        if (t < s) { s1[t] += s1[t + s]; s2[t] += s2[t + s]; }
        __syncthreads();
    }
    if (t == 0) {
        const double hd  = (double)HD_DIM;
        const double nm1 = (double)(N_ROWS - 1);
        const double eps = 1e-8;
        double summu = s1[0], mumu = s2[0];
        double s_scalar = (double)scal[0] / (nm1 * hd * hd);
        // det(eps*I + s*11^T - mu mu^T) = eps^hd * det2 (matrix determinant lemma, rank 2)
        double det2 = (1.0 + s_scalar * hd / eps) * (1.0 - mumu / eps)
                    + s_scalar * summu * summu / (eps * eps);
        double logabsdet = hd * log(eps) + log(fabs(det2));
        double loss1 = sqrt((double)scal[1]) / nm1 + 1.0 * sqrt((double)scal[2]) / nm1;
        double trS = hd * s_scalar - mumu;
        double loss = loss1 + 0.5 * 1.0 * (mumu + trS - hd - logabsdet);
        out[0] = (float)loss1;
        out[1] = (float)loss;
    }
}

// ---------------------------------------------------------------------------
extern "C" void kernel_launch(void* const* d_in, const int* in_sizes, int n_in,
                              void* d_out, int out_size, void* d_ws, size_t ws_size,
                              hipStream_t stream) {
    (void)in_sizes; (void)n_in; (void)out_size; (void)ws_size;
    const float* x  = (const float*)d_in[0];
    const float* h  = (const float*)d_in[1];
    const float* W1 = (const float*)d_in[2];
    const float* b1 = (const float*)d_in[3];
    const float* W2 = (const float*)d_in[4];
    const float* b2 = (const float*)d_in[5];
    float* f = (float*)d_out;                 // [N, P] then loss1, loss

    float* a1    = (float*)d_ws;                           // N*D floats (256 MiB)
    float* musum = a1 + (size_t)N_ROWS * D_DIM;            // [1024]
    float* scal  = musum + HD_DIM;                         // [3]: ssq, sx, sh

    hipLaunchKernelGGL(zero_kernel, dim3(1), dim3(1024), 0, stream, musum, scal);

    // GEMM1: a1 = relu([x|h] @ W1^T + b1)   (virtual concat, split at P)
    hipLaunchKernelGGL(gemm_relu_kernel, dim3(D_DIM / 128, N_ROWS / 128), dim3(256), 0, stream,
                       x, h, P_DIM, HD_DIM, P_DIM, W1, b1, a1, D_DIM, D_DIM);
    // GEMM2: f = relu(a1 @ W2^T + b2)
    hipLaunchKernelGGL(gemm_relu_kernel, dim3(P_DIM / 128, N_ROWS / 128), dim3(256), 0, stream,
                       a1, a1, D_DIM, D_DIM, D_DIM, W2, b2, f, D_DIM, P_DIM);

    hipLaunchKernelGGL(colsum_kernel, dim3(HD_DIM / 256, N_ROWS / 1024), dim3(256), 0, stream,
                       h, musum);
    hipLaunchKernelGGL(rowsq_kernel, dim3(N_ROWS / 8), dim3(256), 0, stream, h, scal + 0);
    hipLaunchKernelGGL(diff_kernel, dim3(2048), dim3(256), 0, stream, f, x, h, scal + 1, scal + 2);
    hipLaunchKernelGGL(finalize_kernel, dim3(1), dim3(1024), 0, stream,
                       musum, scal, f + (size_t)N_ROWS * P_DIM);
}
